// TransformerEncoderLayer_79285096284560
// MI455X (gfx1250) — matrix operand order repaired and checked by reading, not compiled, not run on hardware
//
#include <hip/hip_runtime.h>
#include <hip/hip_bf16.h>
#include <stdint.h>

// ---------------------------------------------------------------------------
// Problem constants (from reference): D=512 H=8 HD=64 E=8 K=2 DFF=1024 B=8 N=256
// Nt = 768, tokens M = B*Nt = 6144
// ---------------------------------------------------------------------------
#define Dm   512
#define Hh   8
#define HD   64
#define Ee   8
#define DFF  1024
#define Bb   8
#define Nt   768
#define Mtok (Bb * Nt)   // 6144

typedef __attribute__((ext_vector_type(8)))  __bf16 v8bf;
typedef __attribute__((ext_vector_type(16))) __bf16 v16bf;
typedef __attribute__((ext_vector_type(8)))  float  v8f;

// ---------------------------------------------------------------------------
// WMMA helpers (CDNA5: D = A(16x32 bf16) * B(32x16 bf16) + C(16x16 f32))
// ---------------------------------------------------------------------------
__device__ __forceinline__ v8f wmma_bf16(v16bf a, v16bf b, v8f c) {
  return __builtin_amdgcn_wmma_f32_16x16x32_bf16(
      /*neg_a=*/false, a, /*neg_b=*/false, b,
      /*c_mod=*/(short)0, c, /*reuse_a=*/false, /*reuse_b=*/false);
}

// A operand: row-major [16][K] bf16 source. Lane l<16 holds row=l, K {k0..k0+7, k0+16..k0+23};
// lane l>=16 holds row=l-16, K {k0+8..k0+15, k0+24..k0+31}.  (ISA 7.12.2, 16-bit A 16x32)
__device__ __forceinline__ v16bf load_a16(const __bf16* base, int lda, int k0) {
  const int lane = threadIdx.x & 31;
  const int row  = lane & 15;
  const int off  = (lane < 16) ? 0 : 8;
  const __bf16* p = base + (size_t)row * lda + k0 + off;
  v8bf lo = *reinterpret_cast<const v8bf*>(p);
  v8bf hi = *reinterpret_cast<const v8bf*>(p + 16);
  return __builtin_shufflevector(lo, hi, 0,1,2,3,4,5,6,7,8,9,10,11,12,13,14,15);
}

// B operand: source is W^T row-major [N][K] bf16 (same K order as A).
// Lane l<16 = column l, K {k0..k0+7, k0+16..k0+23}; lane l>=16 = column l-16, K {k0+8..k0+15, k0+24..k0+31}.
__device__ __forceinline__ v16bf load_b16(const __bf16* bt, int ldb, int k0) {
  const int lane = threadIdx.x & 31;
  const int col  = lane & 15;
  const int off  = (lane < 16) ? 0 : 8;
  const __bf16* p = bt + (size_t)col * ldb + k0 + off;
  v8bf lo = *reinterpret_cast<const v8bf*>(p);
  v8bf hi = *reinterpret_cast<const v8bf*>(p + 16);
  return __builtin_shufflevector(lo, hi, 0,1,2,3,4,5,6,7,8,9,10,11,12,13,14,15);
}

// ---------------------------------------------------------------------------
// Weight transpose + f32 -> bf16.  src [z][K][N] -> dst [z][N][K]
// ---------------------------------------------------------------------------
__global__ void transpose_bf16_kernel(const float* __restrict__ src,
                                      __bf16* __restrict__ dst, int K, int N) {
  const float* s = src + (size_t)blockIdx.z * K * N;
  __bf16* d = dst + (size_t)blockIdx.z * K * N;
  int i = blockIdx.x * 256 + threadIdx.x;
  if (i < K * N) {
    int k = i / N, n = i % N;
    d[(size_t)n * K + k] = (__bf16)s[i];
  }
}

// ---------------------------------------------------------------------------
// LN1 over the (virtually) concatenated input -> bf16 activations [Mtok][512]
// ---------------------------------------------------------------------------
__global__ __launch_bounds__(256) void ln1_kernel(
    const float* __restrict__ x1, const float* __restrict__ x2,
    const float* __restrict__ x3, const float* __restrict__ w,
    const float* __restrict__ b, __bf16* __restrict__ hb) {
  __shared__ float red[256];
  const int t = blockIdx.x;
  const int bb = t / Nt, n = t % Nt;
  const float* src = (n < 256) ? x1 + ((size_t)bb * 256 + n) * Dm
                   : (n < 512) ? x2 + ((size_t)bb * 256 + (n - 256)) * Dm
                               : x3 + ((size_t)bb * 256 + (n - 512)) * Dm;
  float v0 = src[threadIdx.x], v1 = src[threadIdx.x + 256];
  red[threadIdx.x] = v0 + v1; __syncthreads();
  for (int o = 128; o; o >>= 1) { if (threadIdx.x < o) red[threadIdx.x] += red[threadIdx.x + o]; __syncthreads(); }
  float mean = red[0] * (1.f / Dm); __syncthreads();
  float d0 = v0 - mean, d1 = v1 - mean;
  red[threadIdx.x] = d0 * d0 + d1 * d1; __syncthreads();
  for (int o = 128; o; o >>= 1) { if (threadIdx.x < o) red[threadIdx.x] += red[threadIdx.x + o]; __syncthreads(); }
  float inv = rsqrtf(red[0] * (1.f / Dm) + 1e-5f);
  hb[(size_t)t * Dm + threadIdx.x]       = (__bf16)(d0 * inv * w[threadIdx.x] + b[threadIdx.x]);
  hb[(size_t)t * Dm + threadIdx.x + 256] = (__bf16)(d1 * inv * w[threadIdx.x + 256] + b[threadIdx.x + 256]);
}

// ---------------------------------------------------------------------------
// Generic WMMA GEMM: C[M][N] = scale * (A[M][K] @ Bt[N][K]^T + bias)
// block = 256 threads (8 waves); block tile 16 (M) x 512 (N); wave: 4 n-tiles.
// K loop is ping-pong software-pipelined (prefetch k+32 while computing k).
// ---------------------------------------------------------------------------
__global__ __launch_bounds__(256) void gemm_bf16_kernel(
    const __bf16* __restrict__ A, const __bf16* __restrict__ Bt,
    int M, int N, int K, const float* __restrict__ bias, float scale,
    float* __restrict__ Cf, __bf16* __restrict__ Cbf) {
  const int m0   = blockIdx.x * 16;
  const int wave = threadIdx.x >> 5;
  const int n0   = blockIdx.y * 512 + wave * 64;
  if (m0 >= M || n0 >= N) return;
  const __bf16* Arow = A + (size_t)m0 * K;
  v8f acc[4] = {};
  v16bf a0 = load_a16(Arow, K, 0);
  v16bf b0[4], b1[4];
#pragma unroll
  for (int j = 0; j < 4; ++j) b0[j] = load_b16(Bt + (size_t)(n0 + 16 * j) * K, K, 0);
  for (int k0 = 0; k0 < K; k0 += 64) {
    if (k0 + 128 < K) __builtin_prefetch(Bt + (size_t)n0 * K + k0 + 128, 0, 0);
    v16bf a1 = load_a16(Arow, K, k0 + 32);
#pragma unroll
    for (int j = 0; j < 4; ++j) b1[j] = load_b16(Bt + (size_t)(n0 + 16 * j) * K, K, k0 + 32);
#pragma unroll
    for (int j = 0; j < 4; ++j) acc[j] = wmma_bf16(a0, b0[j], acc[j]);
    if (k0 + 64 < K) {
      a0 = load_a16(Arow, K, k0 + 64);
#pragma unroll
      for (int j = 0; j < 4; ++j) b0[j] = load_b16(Bt + (size_t)(n0 + 16 * j) * K, K, k0 + 64);
    }
#pragma unroll
    for (int j = 0; j < 4; ++j) acc[j] = wmma_bf16(a1, b1[j], acc[j]);
  }
  const int lane = threadIdx.x & 31;
  const int cL = lane & 15;
  const int r0 = (lane < 16) ? 0 : 8;
#pragma unroll
  for (int j = 0; j < 4; ++j) {
    int col = n0 + 16 * j + cL;
    float bv = bias ? bias[col] : 0.f;
#pragma unroll
    for (int i = 0; i < 8; ++i) {
      float v = (acc[j][i] + bv) * scale;
      size_t idx = (size_t)(m0 + r0 + i) * N + col;
      if (Cf)  Cf[idx]  = v;
      if (Cbf) Cbf[idx] = (__bf16)v;
    }
  }
}

// ---------------------------------------------------------------------------
// Repack V (cols 512..1023 of kv) into vt[b][h][d][m] for B-operand loads
// ---------------------------------------------------------------------------
__global__ void repack_vt_kernel(const __bf16* __restrict__ kvb, __bf16* __restrict__ vt) {
  int i = blockIdx.x * 256 + threadIdx.x;
  if (i >= Mtok * Dm) return;
  int t = i >> 9, c = i & 511;
  int bb = t / Nt, m = t % Nt;
  int h = c >> 6, d = c & 63;
  vt[(((size_t)(bb * Hh + h)) * HD + d) * Nt + m] = kvb[(size_t)t * 1024 + 512 + c];
}

// ---------------------------------------------------------------------------
// Attention: block = (q-tile of 16, batch-head). 4 waves. Dynamic LDS:
//   sc (f32 scores, 48 KB) + pr (bf16 probs, 24 KB) = 72 KB (CDNA5: 320 KB/WG)
// scores = q @ k^T * 0.125 (WMMA, pipelined) -> softmax -> probs @ v^T (WMMA)
// ---------------------------------------------------------------------------
__global__ __launch_bounds__(128) void attn_kernel(
    const __bf16* __restrict__ qb, const __bf16* __restrict__ kvb,
    const __bf16* __restrict__ vt, __bf16* __restrict__ ob) {
  extern __shared__ __align__(16) unsigned char dynsmem[];
  float*  sc = (float*)dynsmem;                       // [16][768]
  __bf16* pr = (__bf16*)(dynsmem + 16 * Nt * 4);      // [16][768]
  const int qt = blockIdx.x;               // 0..47
  const int bh = blockIdx.y;               // 0..63
  const int bb = bh >> 3, h = bh & 7;
  const int q0 = qt * 16;
  const int wave = threadIdx.x >> 5;
  const int lane = threadIdx.x & 31;
  const int cL = lane & 15;
  const int r0 = (lane < 16) ? 0 : 8;

  const __bf16* Aq    = qb  + ((size_t)(bb * Nt + q0)) * Dm + h * HD;
  const __bf16* Kbase = kvb + ((size_t)(bb * Nt)) * 1024 + h * HD;
  v16bf a0 = load_a16(Aq, Dm, 0);
  v16bf a1 = load_a16(Aq, Dm, 32);
  // scores, ping-pong over this wave's 12 key tiles
  {
    const __bf16* kb = Kbase + (size_t)(wave * 12 * 16) * 1024;
    v16bf c0 = load_b16(kb, 1024, 0);
    v16bf c1 = load_b16(kb, 1024, 32);
    for (int j = 0; j < 12; ++j) {
      v16bf n0v, n1v;
      if (j + 1 < 12) {
        const __bf16* kn = Kbase + (size_t)((wave * 12 + j + 1) * 16) * 1024;
        n0v = load_b16(kn, 1024, 0);
        n1v = load_b16(kn, 1024, 32);
      }
      v8f s = {};
      s = wmma_bf16(a0, c0, s);
      s = wmma_bf16(a1, c1, s);
      int n0 = (wave * 12 + j) * 16;
#pragma unroll
      for (int i = 0; i < 8; ++i) sc[(r0 + i) * Nt + n0 + cL] = s[i] * 0.125f;
      c0 = n0v; c1 = n1v;
    }
  }
  __syncthreads();
  if (threadIdx.x < 16) {                  // row-wise softmax (16 rows)
    int r = threadIdx.x;
    float mx = -1e30f;
    for (int m = 0; m < Nt; ++m) mx = fmaxf(mx, sc[r * Nt + m]);
    float sum = 0.f;
    for (int m = 0; m < Nt; ++m) { float e = __expf(sc[r * Nt + m] - mx); sc[r * Nt + m] = e; sum += e; }
    float inv = 1.f / sum;
    for (int m = 0; m < Nt; ++m) pr[r * Nt + m] = (__bf16)(sc[r * Nt + m] * inv);
  }
  __syncthreads();
  const int d0 = wave * 16;                // 4 waves x 16 = 64 head dims
  const __bf16* Vt = vt + ((size_t)bh * HD + d0) * Nt;
  v8f o = {};
  v16bf pa = load_a16(pr, Nt, 0);
  v16bf vb = load_b16(Vt, Nt, 0);
  for (int m0 = 0; m0 < Nt; m0 += 64) {
    v16bf pa1 = load_a16(pr, Nt, m0 + 32);
    v16bf vb1 = load_b16(Vt, Nt, m0 + 32);
    o = wmma_bf16(pa, vb, o);
    if (m0 + 64 < Nt) {
      pa = load_a16(pr, Nt, m0 + 64);
      vb = load_b16(Vt, Nt, m0 + 64);
    }
    o = wmma_bf16(pa1, vb1, o);
  }
#pragma unroll
  for (int i = 0; i < 8; ++i)
    ob[(size_t)(bb * Nt + q0 + r0 + i) * Dm + h * HD + d0 + cL] = (__bf16)o[i];
}

// ---------------------------------------------------------------------------
// LN2 + gate logits + softmax + top-2 routing per token
// gate[t] = {conf0, conf1, (float)idx0, (float)idx1}
// ---------------------------------------------------------------------------
__global__ __launch_bounds__(256) void ln2_gate_kernel(
    const float* __restrict__ xf, const float* __restrict__ w, const float* __restrict__ b,
    const float* __restrict__ gw, __bf16* __restrict__ h2b, float* __restrict__ gate) {
  __shared__ float red[256];
  __shared__ float row[512];
  __shared__ float lg[8];
  const int t = blockIdx.x;
  const float* x = xf + (size_t)t * Dm;
  float v0 = x[threadIdx.x], v1 = x[threadIdx.x + 256];
  red[threadIdx.x] = v0 + v1; __syncthreads();
  for (int o = 128; o; o >>= 1) { if (threadIdx.x < o) red[threadIdx.x] += red[threadIdx.x + o]; __syncthreads(); }
  float mean = red[0] * (1.f / Dm); __syncthreads();
  float d0 = v0 - mean, d1 = v1 - mean;
  red[threadIdx.x] = d0 * d0 + d1 * d1; __syncthreads();
  for (int o = 128; o; o >>= 1) { if (threadIdx.x < o) red[threadIdx.x] += red[threadIdx.x + o]; __syncthreads(); }
  float inv = rsqrtf(red[0] * (1.f / Dm) + 1e-5f);
  float h0 = d0 * inv * w[threadIdx.x] + b[threadIdx.x];
  float h1 = d1 * inv * w[threadIdx.x + 256] + b[threadIdx.x + 256];
  row[threadIdx.x] = h0; row[threadIdx.x + 256] = h1;
  h2b[(size_t)t * Dm + threadIdx.x]       = (__bf16)h0;
  h2b[(size_t)t * Dm + threadIdx.x + 256] = (__bf16)h1;
  __syncthreads();
  if (threadIdx.x < 8) {
    float acc = 0.f;
    for (int i = 0; i < Dm; ++i) acc += row[i] * gw[i * Ee + threadIdx.x];
    lg[threadIdx.x] = acc;
  }
  __syncthreads();
  if (threadIdx.x == 0) {
    float mx = lg[0];
    for (int e = 1; e < Ee; ++e) mx = fmaxf(mx, lg[e]);
    float p[Ee], s = 0.f;
    for (int e = 0; e < Ee; ++e) { p[e] = __expf(lg[e] - mx); s += p[e]; }
    for (int e = 0; e < Ee; ++e) p[e] /= s;
    int i0 = 0;
    for (int e = 1; e < Ee; ++e) if (p[e] > p[i0]) i0 = e;
    int i1 = (i0 == 0) ? 1 : 0;
    for (int e = 0; e < Ee; ++e) if (e != i0 && p[e] > p[i1]) i1 = e;
    float ss = p[i0] + p[i1];
    float* g = gate + (size_t)t * 4;
    g[0] = p[i0] / ss; g[1] = p[i1] / ss; g[2] = (float)i0; g[3] = (float)i1;
  }
}

// ---------------------------------------------------------------------------
// Fused MoE + final LN. One block per 16-token tile (8 waves).
// Loops experts; skips experts unused by the tile; hid tile in LDS;
// gated accumulation in registers; final residual LayerNorm fused.
// Both GEMM phases ping-pong pipelined (phase 1 in two 4-tile halves to
// bound VGPR pressure at ~144).
// ---------------------------------------------------------------------------
__global__ __launch_bounds__(256) void moe_final_kernel(
    const __bf16* __restrict__ h2b, const float* __restrict__ xf,
    const __bf16* __restrict__ ew1t, const float* __restrict__ eb1,
    const __bf16* __restrict__ ew2t, const float* __restrict__ eb2,
    const float* __restrict__ gate, const float* __restrict__ lnw,
    const float* __restrict__ lnb, float* __restrict__ out) {
  __shared__ __align__(16) unsigned char smem[32768 + 16384 + 4096];
  __bf16* hid   = (__bf16*)smem;                    // [16][1024] per-expert hidden
  float*  combf = (float*)smem;                     // [16][512]  (reused after expert loop)
  __bf16* h2s   = (__bf16*)(smem + 32768);          // [16][512]  staged LN2 activations
  float*  wrow  = (float*)(smem + 32768 + 16384);   // [16] gate weights
  float*  redA  = wrow + 16;                        // [16][17]
  float*  redB  = redA + 16 * 17;                   // [16][17]
  float*  rstat = redB + 16 * 17;                   // [16][2]

  const int t0   = blockIdx.x * 16;
  const int wave = threadIdx.x >> 5;
  const int lane = threadIdx.x & 31;
  const int cL   = lane & 15;
  const int r0   = (lane < 16) ? 0 : 8;

  { // stage h2 tile into LDS (16 KB)
    const uint32_t* s = (const uint32_t*)(h2b + (size_t)t0 * Dm);
    uint32_t* d = (uint32_t*)h2s;
    for (int i = threadIdx.x; i < 16 * Dm / 2; i += 256) d[i] = s[i];
  }
  v8f acc_out[4] = {};
  __syncthreads();

  for (int e = 0; e < Ee; ++e) {
    if (threadIdx.x < 16) {
      const float* g = gate + (size_t)(t0 + threadIdx.x) * 4;
      float w = 0.f;
      if ((int)g[2] == e)      w = g[0];
      else if ((int)g[3] == e) w = g[1];
      wrow[threadIdx.x] = w;
    }
    __syncthreads();
    float ssum = 0.f;
    for (int r = 0; r < 16; ++r) ssum += wrow[r];
    if (ssum > 0.f) {
      // ---- phase 1: hid = gelu(h2 @ w1[e] + b1[e]) ----
      // wave covers 8 n-tiles as two pipelined halves of 4
#pragma unroll
      for (int half = 0; half < 2; ++half) {
        const __bf16* btb = ew1t + ((size_t)e * DFF + wave * 128 + half * 64) * Dm;
        v8f acc1[4] = {};
        v16bf a0 = load_a16(h2s, Dm, 0);
        v16bf b0[4], b1v[4];
#pragma unroll
        for (int j = 0; j < 4; ++j) b0[j] = load_b16(btb + (size_t)(16 * j) * Dm, Dm, 0);
        for (int k0 = 0; k0 < Dm; k0 += 64) {
          if (k0 + 128 < Dm) __builtin_prefetch(btb + k0 + 128, 0, 0);
          v16bf a1 = load_a16(h2s, Dm, k0 + 32);
#pragma unroll
          for (int j = 0; j < 4; ++j) b1v[j] = load_b16(btb + (size_t)(16 * j) * Dm, Dm, k0 + 32);
#pragma unroll
          for (int j = 0; j < 4; ++j) acc1[j] = wmma_bf16(a0, b0[j], acc1[j]);
          if (k0 + 64 < Dm) {
            a0 = load_a16(h2s, Dm, k0 + 64);
#pragma unroll
            for (int j = 0; j < 4; ++j) b0[j] = load_b16(btb + (size_t)(16 * j) * Dm, Dm, k0 + 64);
          }
#pragma unroll
          for (int j = 0; j < 4; ++j) acc1[j] = wmma_bf16(a1, b1v[j], acc1[j]);
        }
#pragma unroll
        for (int j = 0; j < 4; ++j) {
          int col = wave * 128 + half * 64 + j * 16 + cL;
          float bv = eb1[e * DFF + col];
#pragma unroll
          for (int i = 0; i < 8; ++i) {
            float v = acc1[j][i] + bv;
            hid[(r0 + i) * DFF + col] = (__bf16)(0.5f * v * (1.f + erff(v * 0.70710678118f)));
          }
        }
      }
      __syncthreads();
      // ---- phase 2: gated accumulate of hid @ w2[e] + b2[e] ---- (4 n-tiles)
      {
        const __bf16* btb = ew2t + ((size_t)e * Dm + wave * 64) * DFF;
        v8f acc2[4] = {};
        v16bf a0 = load_a16(hid, DFF, 0);
        v16bf b0[4], b1v[4];
#pragma unroll
        for (int j = 0; j < 4; ++j) b0[j] = load_b16(btb + (size_t)(16 * j) * DFF, DFF, 0);
        for (int k0 = 0; k0 < DFF; k0 += 64) {
          if (k0 + 128 < DFF) __builtin_prefetch(btb + k0 + 128, 0, 0);
          v16bf a1 = load_a16(hid, DFF, k0 + 32);
#pragma unroll
          for (int j = 0; j < 4; ++j) b1v[j] = load_b16(btb + (size_t)(16 * j) * DFF, DFF, k0 + 32);
#pragma unroll
          for (int j = 0; j < 4; ++j) acc2[j] = wmma_bf16(a0, b0[j], acc2[j]);
          if (k0 + 64 < DFF) {
            a0 = load_a16(hid, DFF, k0 + 64);
#pragma unroll
            for (int j = 0; j < 4; ++j) b0[j] = load_b16(btb + (size_t)(16 * j) * DFF, DFF, k0 + 64);
          }
#pragma unroll
          for (int j = 0; j < 4; ++j) acc2[j] = wmma_bf16(a1, b1v[j], acc2[j]);
        }
#pragma unroll
        for (int j = 0; j < 4; ++j) {
          int col = wave * 64 + j * 16 + cL;
          float bv = eb2[e * Dm + col];
#pragma unroll
          for (int i = 0; i < 8; ++i)
            acc_out[j][i] += wrow[r0 + i] * (acc2[j][i] + bv);
        }
      }
    }
    __syncthreads();  // protect hid / wrow reuse across experts
  }

  // spill combined tile to LDS (overlays hid, now dead)
#pragma unroll
  for (int j = 0; j < 4; ++j) {
    int col = wave * 64 + j * 16 + cL;
#pragma unroll
    for (int i = 0; i < 8; ++i) combf[(r0 + i) * Dm + col] = acc_out[j][i];
  }
  __syncthreads();

  // fused final LN: out = LN(x + comb) ; 16 threads per row
  {
    const int r = threadIdx.x >> 4;
    const int l = threadIdx.x & 15;
    const float* xr = xf + (size_t)(t0 + r) * Dm;
    float s = 0.f, q = 0.f;
    for (int i = l; i < Dm; i += 16) {
      float v = xr[i] + combf[r * Dm + i];
      s += v; q += v * v;
    }
    redA[r * 17 + l] = s; redB[r * 17 + l] = q;
    __syncthreads();
    if (l == 0) {
      float ts = 0.f, tq = 0.f;
      for (int i = 0; i < 16; ++i) { ts += redA[r * 17 + i]; tq += redB[r * 17 + i]; }
      float mean = ts * (1.f / Dm);
      float var  = tq * (1.f / Dm) - mean * mean;
      rstat[r * 2] = mean; rstat[r * 2 + 1] = rsqrtf(var + 1e-5f);
    }
    __syncthreads();
    float mean = rstat[r * 2], inv = rstat[r * 2 + 1];
    float* orow = out + (size_t)(t0 + r) * Dm;
    for (int i = l; i < Dm; i += 16) {
      float v = xr[i] + combf[r * Dm + i];
      orow[i] = (v - mean) * inv * lnw[i] + lnb[i];
    }
  }
}

// ---------------------------------------------------------------------------
// Host-side launch
// ---------------------------------------------------------------------------
extern "C" void kernel_launch(void* const* d_in, const int* in_sizes, int n_in,
                              void* d_out, int out_size, void* d_ws, size_t ws_size,
                              hipStream_t stream) {
  const float* x1      = (const float*)d_in[0];
  const float* x2      = (const float*)d_in[1];
  const float* x3      = (const float*)d_in[2];
  const float* norm1_w = (const float*)d_in[3];
  const float* norm1_b = (const float*)d_in[4];
  const float* norm2_w = (const float*)d_in[5];
  const float* norm2_b = (const float*)d_in[6];
  const float* ln1_w   = (const float*)d_in[7];
  const float* ln1_b   = (const float*)d_in[8];
  const float* q_w     = (const float*)d_in[9];
  const float* kv_w    = (const float*)d_in[10];
  const float* proj_w  = (const float*)d_in[11];
  const float* proj_b  = (const float*)d_in[12];
  const float* gate_w  = (const float*)d_in[13];
  const float* e_w1    = (const float*)d_in[14];
  const float* e_b1    = (const float*)d_in[15];
  const float* e_w2    = (const float*)d_in[16];
  const float* e_b2    = (const float*)d_in[17];
  float* outp = (float*)d_out;

  char* ws = (char*)d_ws;
  auto alloc = [&](size_t bytes) -> char* {
    char* p = ws; ws += (bytes + 255) & ~(size_t)255; return p;
  };
  __bf16* qwt   = (__bf16*)alloc((size_t)Dm * Dm * 2);
  __bf16* kvwt  = (__bf16*)alloc((size_t)Dm * 2 * Dm * 2);
  __bf16* pwt   = (__bf16*)alloc((size_t)Dm * Dm * 2);
  __bf16* ew1t  = (__bf16*)alloc((size_t)Ee * DFF * Dm * 2);
  __bf16* ew2t  = (__bf16*)alloc((size_t)Ee * Dm * DFF * 2);
  __bf16* h_bf  = (__bf16*)alloc((size_t)Mtok * Dm * 2);
  __bf16* q_bf  = (__bf16*)alloc((size_t)Mtok * Dm * 2);
  __bf16* kv_bf = (__bf16*)alloc((size_t)Mtok * 2 * Dm * 2);
  __bf16* vt    = (__bf16*)alloc((size_t)Mtok * Dm * 2);
  __bf16* o_bf  = (__bf16*)alloc((size_t)Mtok * Dm * 2);
  float*  x_f   = (float*) alloc((size_t)Mtok * Dm * 4);
  __bf16* h2_bf = (__bf16*)alloc((size_t)Mtok * Dm * 2);
  float*  gate  = (float*) alloc((size_t)Mtok * 4 * 4);
  (void)ws_size; (void)in_sizes; (void)n_in; (void)out_size;

  // 1. weights -> bf16, transposed to [N][K]
  transpose_bf16_kernel<<<dim3((Dm * Dm + 255) / 256, 1, 1), 256, 0, stream>>>(q_w, qwt, Dm, Dm);
  transpose_bf16_kernel<<<dim3((Dm * 2 * Dm + 255) / 256, 1, 1), 256, 0, stream>>>(kv_w, kvwt, Dm, 2 * Dm);
  transpose_bf16_kernel<<<dim3((Dm * Dm + 255) / 256, 1, 1), 256, 0, stream>>>(proj_w, pwt, Dm, Dm);
  transpose_bf16_kernel<<<dim3((Dm * DFF + 255) / 256, 1, Ee), 256, 0, stream>>>(e_w1, ew1t, Dm, DFF);
  transpose_bf16_kernel<<<dim3((DFF * Dm + 255) / 256, 1, Ee), 256, 0, stream>>>(e_w2, ew2t, DFF, Dm);

  // 2. LN1 -> bf16 activations
  ln1_kernel<<<Mtok, 256, 0, stream>>>(x1, x2, x3, norm1_w, norm1_b, h_bf);

  // 3. Q and KV projections (WMMA)
  gemm_bf16_kernel<<<dim3(Mtok / 16, 1), 256, 0, stream>>>(
      h_bf, qwt, Mtok, Dm, Dm, nullptr, 1.f, nullptr, q_bf);
  gemm_bf16_kernel<<<dim3(Mtok / 16, 2), 256, 0, stream>>>(
      h_bf, kvwt, Mtok, 2 * Dm, Dm, nullptr, 1.f, nullptr, kv_bf);

  // 4. repack V transposed per (b,h) for B-operand loads
  repack_vt_kernel<<<(Mtok * Dm + 255) / 256, 256, 0, stream>>>(kv_bf, vt);

  // 5. attention (WMMA scores + softmax + WMMA PV), 72 KB dynamic LDS
  attn_kernel<<<dim3(Nt / 16, Bb * Hh), 128, 16 * Nt * 4 + 16 * Nt * 2, stream>>>(
      q_bf, kv_bf, vt, o_bf);

  // 6. output projection, x = 2*(o @ proj_w + proj_b)
  gemm_bf16_kernel<<<dim3(Mtok / 16, 1), 256, 0, stream>>>(
      o_bf, pwt, Mtok, Dm, Dm, proj_b, 2.f, x_f, nullptr);

  // 7. LN2 + gate softmax + top-2
  ln2_gate_kernel<<<Mtok, 256, 0, stream>>>(x_f, norm2_w, norm2_b, gate_w, h2_bf, gate);

  // 8. fused MoE (expert-skipping, WMMA) + final LayerNorm
  moe_final_kernel<<<Mtok / 16, 256, 0, stream>>>(
      h2_bf, x_f, ew1t, e_b1, ew2t, e_b2, gate, ln1_w, ln1_b, outp);
}
